// CosineAttention_28054726377595
// MI455X (gfx1250) — matrix operand order repaired and checked by reading, not compiled, not run on hardware
//
#include <hip/hip_runtime.h>
#include <hip/hip_bf16.h>
#include <stdint.h>

// ---------------------------------------------------------------------------
// CDNA5 / gfx1250 fused cosine-graph-attention.
//  - bf16 WMMA (v_wmma_f32_16x16x32_bf16) for all three matmuls, f32 accum.
//  - flash-attention style online softmax: sim (8192x8192) never materialized.
//  - Wh kept as bf16 row-major + bf16 transposed copy (8 MB each, L2-resident).
//  - M_TILE = 32 query rows per block: every K/V B-operand load feeds two
//    WMMAs, and total L2 re-read traffic is halved vs a 16-row tile.
// ---------------------------------------------------------------------------

typedef __attribute__((ext_vector_type(16))) __bf16       v16bf;
typedef __attribute__((ext_vector_type(8)))  float        v8f;
typedef __attribute__((ext_vector_type(4)))  unsigned int u32x4;

union BF16Tile { v16bf v; u32x4 q[2]; };

#define N_NODES 8192
#define DIM     512
#define EPS     1e-8f
#define M_TILE  32

__device__ __forceinline__ unsigned short f32_to_bf16(float f) {
  unsigned u = __float_as_uint(f);
  u += 0x7FFFu + ((u >> 16) & 1u);          // round to nearest even
  return (unsigned short)(u >> 16);
}

// ---------------------------------------------------------------------------
// Kernel 1: f32 -> bf16 conversion of h and W, zero norm accumulators.
// ---------------------------------------------------------------------------
__global__ __launch_bounds__(256) void prep_kernel(
    const float* __restrict__ h, const float* __restrict__ W,
    unsigned short* __restrict__ h_bf, unsigned short* __restrict__ W_bf,
    float* __restrict__ normsq) {
  int i = blockIdx.x * 256 + threadIdx.x;
  if (i < N_NODES * DIM) h_bf[i] = f32_to_bf16(h[i]);
  if (i < DIM * DIM)     W_bf[i] = f32_to_bf16(W[i]);
  if (i < N_NODES)       normsq[i] = 0.0f;
}

// ---------------------------------------------------------------------------
// Kernel 2: Wh = h @ W^T   (8192x512 = [8192x512]x[512x512])
// 512 blocks x 8 waves; wave w -> 16 rows x 64 outs (4 WMMA accumulators).
// Emits Wh in bf16 (row major), bf16 transpose, and row sum-of-squares.
// ---------------------------------------------------------------------------
__global__ __launch_bounds__(256) void gemm_wh_kernel(
    const unsigned short* __restrict__ h_bf, const unsigned short* __restrict__ W_bf,
    unsigned short* __restrict__ Wh_bf, unsigned short* __restrict__ WhT_bf,
    float* __restrict__ normsq) {
  const int m0   = blockIdx.x * 16;
  const int wave = threadIdx.x >> 5;
  const int lane = threadIdx.x & 31;
  const int lm   = lane & 15;         // column / row-within-16
  const int half = lane >> 4;         // K-half selector
  const int o0   = wave * 64;

  v8f acc[4] = {};
  const unsigned short* arow = h_bf + (size_t)(m0 + lm) * DIM;

  #pragma unroll
  for (int c = 0; c < DIM / 32; ++c) {
    BF16Tile a;
    a.q[0] = *(const u32x4*)(arow + c * 32 + 16 * half);       // K 16h..16h+7
    a.q[1] = *(const u32x4*)(arow + c * 32 + 16 * half + 8);   // K 16h+8..
    #pragma unroll
    for (int nt = 0; nt < 4; ++nt) {
      const unsigned short* brow =
          W_bf + (size_t)(o0 + nt * 16 + lm) * DIM + c * 32 + 16 * half;
      BF16Tile b;
      b.q[0] = *(const u32x4*)(brow);
      b.q[1] = *(const u32x4*)(brow + 8);
      acc[nt] = __builtin_amdgcn_wmma_f32_16x16x32_bf16(
          false, a.v, false, b.v, (short)0, acc[nt], false, false);
    }
  }

  // Write bf16 Wh + transpose, accumulate row sum-of-squares.
  #pragma unroll
  for (int r = 0; r < 8; ++r) {
    const int row = m0 + r + 8 * half;
    float ss = 0.0f;
    #pragma unroll
    for (int nt = 0; nt < 4; ++nt) {
      const float v   = acc[nt][r];
      const int   col = o0 + nt * 16 + lm;
      ss += v * v;
      Wh_bf [(size_t)row * DIM     + col] = f32_to_bf16(v);
      WhT_bf[(size_t)col * N_NODES + row] = f32_to_bf16(v);
    }
    ss += __shfl_xor(ss, 1);
    ss += __shfl_xor(ss, 2);
    ss += __shfl_xor(ss, 4);
    ss += __shfl_xor(ss, 8);
    if (lm == 0) atomicAdd(&normsq[row], ss);   // global_atomic_add_f32
  }
}

// ---------------------------------------------------------------------------
// Kernel 3: norms = sqrt(normsq)
// ---------------------------------------------------------------------------
__global__ __launch_bounds__(256) void norm_kernel(
    const float* __restrict__ normsq, float* __restrict__ norms) {
  int i = blockIdx.x * 256 + threadIdx.x;
  if (i < N_NODES) norms[i] = sqrtf(normsq[i]);
}

// ---------------------------------------------------------------------------
// Kernel 4: fused flash attention, M_TILE = 32 query rows per block.
// 8 waves. Per 128-column step:
//   P1: wave w computes two 16x16 S tiles (Q rows 0-15 / 16-31 vs its 16 key
//       rows); each K-chunk B operand is loaded once and feeds 2 WMMAs.
//   P2: 256 threads (32 rows x 8 threads/row): online softmax, P -> bf16 LDS.
//   P3: wave w rescales its 32x64 f32 accumulators and does P x V; each V
//       B operand feeds 2 WMMAs.
// ---------------------------------------------------------------------------
__global__ __launch_bounds__(256) void attn_kernel(
    const unsigned short* __restrict__ Wh_bf,
    const unsigned short* __restrict__ WhT_bf,
    const float* __restrict__ norms, const int* __restrict__ adj,
    float* __restrict__ out) {
  __shared__ unsigned short Qs[M_TILE * DIM];     // 32 KB   Q tile bf16
  __shared__ float          Ss[M_TILE * 128];     // 16 KB   sim tile f32
  __shared__ unsigned short Ps[M_TILE * 128];     //  8 KB   P tile bf16
  __shared__ float m_s[M_TILE], l_s[M_TILE], corr_s[M_TILE];

  const float NEGINF = -__builtin_inff();
  const int m0   = blockIdx.x * M_TILE;
  const int tid  = threadIdx.x;
  const int wave = tid >> 5;
  const int lane = tid & 31;
  const int lm   = lane & 15;
  const int half = lane >> 4;

  // Stage Q tile (Wh rows m0..m0+31, bf16) into LDS.
  {
    const unsigned int* qsrc = (const unsigned int*)(Wh_bf + (size_t)m0 * DIM);
    unsigned int* qdst = (unsigned int*)Qs;
    #pragma unroll
    for (int i = 0; i < M_TILE * DIM / 2 / 256; ++i)
      qdst[tid + i * 256] = qsrc[tid + i * 256];
  }
  if (tid < M_TILE) { m_s[tid] = NEGINF; l_s[tid] = 0.0f; }

  // Per-lane query norms for the 2x8 accumulator rows.
  float qn[2][8];
  #pragma unroll
  for (int mh = 0; mh < 2; ++mh)
    #pragma unroll
    for (int r = 0; r < 8; ++r)
      qn[mh][r] = norms[m0 + mh * 16 + r + 8 * half];

  v8f acc[2][4] = {};
  __syncthreads();

  for (int j0 = 0; j0 < N_NODES; j0 += 128) {
    const int jb = j0 + 16 * wave;            // this wave's 16 key rows

    // ---- Phase 1: S = Q x K^T (two 16x16 tiles share each B load) ------
    v8f s[2] = {};
    const unsigned short* krow = Wh_bf + (size_t)(jb + lm) * DIM + 16 * half;
    #pragma unroll
    for (int c = 0; c < DIM / 32; ++c) {
      BF16Tile b;
      b.q[0] = *(const u32x4*)(krow + c * 32);                  // global b128
      b.q[1] = *(const u32x4*)(krow + c * 32 + 8);
      #pragma unroll
      for (int mh = 0; mh < 2; ++mh) {
        const unsigned short* qrow = Qs + (mh * 16 + lm) * DIM;
        BF16Tile a;
        a.q[0] = *(const u32x4*)(qrow + c * 32 + 16 * half);    // ds_load_b128
        a.q[1] = *(const u32x4*)(qrow + c * 32 + 16 * half + 8);
        s[mh] = __builtin_amdgcn_wmma_f32_16x16x32_bf16(
            false, a.v, false, b.v, (short)0, s[mh], false, false);
      }
    }
    const float kn = norms[jb + lm];
    #pragma unroll
    for (int mh = 0; mh < 2; ++mh) {
      #pragma unroll
      for (int r = 0; r < 8; ++r) {
        const int lrow = mh * 16 + r + 8 * half;  // row within tile
        const int gr = m0 + lrow;                 // global query row
        const int gc = jb + lm;                   // global key row
        const float sim = s[mh][r] / (qn[mh][r] * kn + EPS);
        const int av = adj[(size_t)gr * N_NODES + gc];
        const bool keep = (av > 0) || (gr == gc);
        Ss[lrow * 128 + 16 * wave + lm] = keep ? sim : NEGINF;
      }
    }
    __syncthreads();

    // ---- Phase 2: online softmax (32 rows x 8 threads/row) -------------
    {
      const int row = tid >> 3, cid = tid & 7;
      float vals[16];
      float rmax = NEGINF;
      #pragma unroll
      for (int k = 0; k < 16; ++k) {
        vals[k] = Ss[row * 128 + cid + 8 * k];
        rmax = fmaxf(rmax, vals[k]);
      }
      rmax = fmaxf(rmax, __shfl_xor(rmax, 1));
      rmax = fmaxf(rmax, __shfl_xor(rmax, 2));
      rmax = fmaxf(rmax, __shfl_xor(rmax, 4));
      const float mprev = m_s[row];
      const float mnew  = fmaxf(mprev, rmax);
      const float corr  = (mprev == NEGINF) ? 0.0f : __expf(mprev - mnew);
      float lsum = 0.0f;
      #pragma unroll
      for (int k = 0; k < 16; ++k) {
        const float p = (vals[k] == NEGINF) ? 0.0f : __expf(vals[k] - mnew);
        lsum += p;
        Ps[row * 128 + cid + 8 * k] = f32_to_bf16(p);
      }
      lsum += __shfl_xor(lsum, 1);
      lsum += __shfl_xor(lsum, 2);
      lsum += __shfl_xor(lsum, 4);
      if (cid == 0) {
        l_s[row] = corr * l_s[row] + lsum;
        m_s[row] = mnew;
        corr_s[row] = corr;
      }
    }
    __syncthreads();

    // ---- Phase 3: rescale accumulators + P x V -------------------------
    #pragma unroll
    for (int mh = 0; mh < 2; ++mh) {
      #pragma unroll
      for (int r = 0; r < 8; ++r) {
        const float c = corr_s[mh * 16 + r + 8 * half];
        #pragma unroll
        for (int nt = 0; nt < 4; ++nt) acc[mh][nt][r] *= c;
      }
    }
    #pragma unroll
    for (int c4 = 0; c4 < 4; ++c4) {              // K = 128 cols, 32 per WMMA
      BF16Tile a[2];
      #pragma unroll
      for (int mh = 0; mh < 2; ++mh) {
        const unsigned short* prow = Ps + (mh * 16 + lm) * 128;
        a[mh].q[0] = *(const u32x4*)(prow + c4 * 32 + 16 * half);
        a[mh].q[1] = *(const u32x4*)(prow + c4 * 32 + 16 * half + 8);
      }
      #pragma unroll
      for (int nt = 0; nt < 4; ++nt) {
        const unsigned short* vrow =
            WhT_bf + (size_t)(wave * 64 + nt * 16 + lm) * N_NODES +
            j0 + c4 * 32 + 16 * half;
        BF16Tile b;
        b.q[0] = *(const u32x4*)(vrow);
        b.q[1] = *(const u32x4*)(vrow + 8);
        #pragma unroll
        for (int mh = 0; mh < 2; ++mh) {
          acc[mh][nt] = __builtin_amdgcn_wmma_f32_16x16x32_bf16(
              false, a[mh].v, false, b.v, (short)0, acc[mh][nt], false, false);
        }
      }
    }
    // No barrier needed here: next iteration's post-P1 barrier orders the
    // Ps/corr_s reads above against the next phase-2 writes.
  }
  __syncthreads();

  // ---- Epilogue: out = acc / l ----------------------------------------
  #pragma unroll
  for (int mh = 0; mh < 2; ++mh) {
    #pragma unroll
    for (int r = 0; r < 8; ++r) {
      const int lrow = mh * 16 + r + 8 * half;
      const float li = 1.0f / l_s[lrow];
      #pragma unroll
      for (int nt = 0; nt < 4; ++nt) {
        out[(size_t)(m0 + lrow) * DIM + wave * 64 + nt * 16 + lm] =
            acc[mh][nt][r] * li;
      }
    }
  }
}

// ---------------------------------------------------------------------------
// Launch
// ---------------------------------------------------------------------------
extern "C" void kernel_launch(void* const* d_in, const int* in_sizes, int n_in,
                              void* d_out, int out_size, void* d_ws, size_t ws_size,
                              hipStream_t stream) {
  const float* h   = (const float*)d_in[0];   // [8192,512] f32
  const int*   adj = (const int*)d_in[1];     // [8192,8192] i32
  const float* W   = (const float*)d_in[2];   // [512,512] f32
  float* out = (float*)d_out;                 // [8192,512] f32

  char* ws = (char*)d_ws;
  unsigned short* h_bf   = (unsigned short*)(ws + 0);          //  8 MB
  unsigned short* W_bf   = (unsigned short*)(ws + 8388608);    // 512 KB
  unsigned short* Wh_bf  = (unsigned short*)(ws + 8912896);    //  8 MB
  unsigned short* WhT_bf = (unsigned short*)(ws + 17301504);   //  8 MB
  float*          normsq = (float*)(ws + 25690112);            // 32 KB
  float*          norms  = (float*)(ws + 25722880);            // 32 KB

  prep_kernel   <<<(N_NODES * DIM + 255) / 256, 256, 0, stream>>>(h, W, h_bf, W_bf, normsq);
  gemm_wh_kernel<<<N_NODES / 16, 256, 0, stream>>>(h_bf, W_bf, Wh_bf, WhT_bf, normsq);
  norm_kernel   <<<N_NODES / 256, 256, 0, stream>>>(normsq, norms);
  attn_kernel   <<<N_NODES / M_TILE, 256, 0, stream>>>(Wh_bf, WhT_bf, norms, adj, out);
}
